// MultiHeadAttention_3530463117806
// MI455X (gfx1250) — matrix operand order repaired and checked by reading, not compiled, not run on hardware
//
#include <hip/hip_runtime.h>

#define DMODEL 512
#define HEADS  8
#define HD     64
#define SEQ    2048

typedef __attribute__((ext_vector_type(16))) __bf16 bf16x16;
typedef __attribute__((ext_vector_type(8)))  float  f32x8;

union Frag {
  bf16x16 v;
  uint4   q[2];
  __bf16  bf[16];
  unsigned short u[16];
};

static __device__ __forceinline__ unsigned short f2bf(float x) {
  union { __bf16 h; unsigned short u; } cv;
  cv.h = (__bf16)x;          // native v_cvt f32->bf16 path
  return cv.u;
}

static __device__ __forceinline__ f32x8 wmma_bf16(const Frag& a, const Frag& b, f32x8 c) {
  return __builtin_amdgcn_wmma_f32_16x16x32_bf16(false, a.v, false, b.v,
                                                 (short)0, c, false, false);
}

// xor-lane swap within 16-lane rows via v_permlane16_b32 (VALU, no LDS port).
template <unsigned S0, unsigned S1>
static __device__ __forceinline__ float plx(float x) {
  int i = __float_as_int(x);
  int r = __builtin_amdgcn_permlane16(i, i, (int)S0, (int)S1, false, false);
  return __int_as_float(r);
}
static __device__ __forceinline__ float xormax16(float t) {
  t = fmaxf(t, plx<0x67452301u, 0xEFCDAB89u>(t));   // ^1
  t = fmaxf(t, plx<0x54761032u, 0xDCFE98BAu>(t));   // ^2
  t = fmaxf(t, plx<0x32107654u, 0xBA98FEDCu>(t));   // ^4
  t = fmaxf(t, plx<0xFEDCBA98u, 0x76543210u>(t));   // ^8
  return t;
}
static __device__ __forceinline__ float xorsum16(float t) {
  t += plx<0x67452301u, 0xEFCDAB89u>(t);
  t += plx<0x54761032u, 0xDCFE98BAu>(t);
  t += plx<0x32107654u, 0xBA98FEDCu>(t);
  t += plx<0xFEDCBA98u, 0x76543210u>(t);
  return t;
}

// ---------------------------------------------------------------------------
// Kernel 1: convert 4 weight matrices (512x512 fp32, row-major W[n,k]) to bf16
// ---------------------------------------------------------------------------
__global__ __launch_bounds__(256) void wcvt_kernel(
    const float* __restrict__ w0, const float* __restrict__ w1,
    const float* __restrict__ w2, const float* __restrict__ w3,
    unsigned short* __restrict__ dst) {
  const float* src = (blockIdx.z == 0) ? w0 : (blockIdx.z == 1) ? w1
                   : (blockIdx.z == 2) ? w2 : w3;
  unsigned short* out = dst + (size_t)blockIdx.z * DMODEL * DMODEL;
  int idx = (blockIdx.x * 256 + threadIdx.x) * 4;
  float4 f = *(const float4*)(src + idx);
  out[idx + 0] = f2bf(f.x);
  out[idx + 1] = f2bf(f.y);
  out[idx + 2] = f2bf(f.z);
  out[idx + 3] = f2bf(f.w);
}

// ---------------------------------------------------------------------------
// Kernel 2/4: GEMM  C[8192,512] = X[8192,512] @ W^T + bias
//   WG tile 64x128, 8 waves (4 row-groups x 2 col-groups), 4 accums/wave.
// ---------------------------------------------------------------------------
template <int XBF16, int OUTMODE>
__global__ __launch_bounds__(256) void gemm_kernel(
    const void* __restrict__ Xv, const unsigned short* __restrict__ Wb,
    const float* __restrict__ bias, void* __restrict__ Out) {
  const int lane = threadIdx.x & 31;
  const int wv   = threadIdx.x >> 5;
  const int r    = wv & 3;
  const int cw   = wv >> 2;
  const int mb   = blockIdx.x * 64 + r * 16;
  const int nb   = blockIdx.y * 128 + cw * 64;
  const int half = lane >> 4;
  const int l15  = lane & 15;

  const float* Xf = (const float*)Xv;
  const unsigned short* Xh = (const unsigned short*)Xv;

  f32x8 acc[4];
#pragma unroll
  for (int t = 0; t < 4; ++t) acc[t] = (f32x8){};
  const int arow = mb + l15;

#pragma unroll 2
  for (int k = 0; k < DMODEL; k += 32) {
    Frag a;
    const int kc = k + half * 8;
    if (XBF16) {
      const unsigned short* p = Xh + (size_t)arow * DMODEL + kc;
      a.q[0] = *(const uint4*)(p);
      a.q[1] = *(const uint4*)(p + 16);
    } else {
      const float* p = Xf + (size_t)arow * DMODEL + kc;
      float4 f0 = *(const float4*)(p);
      float4 f1 = *(const float4*)(p + 4);
      float4 f2 = *(const float4*)(p + 16);
      float4 f3 = *(const float4*)(p + 20);
      a.bf[0] = (__bf16)f0.x;  a.bf[1] = (__bf16)f0.y;
      a.bf[2] = (__bf16)f0.z;  a.bf[3] = (__bf16)f0.w;
      a.bf[4] = (__bf16)f1.x;  a.bf[5] = (__bf16)f1.y;
      a.bf[6] = (__bf16)f1.z;  a.bf[7] = (__bf16)f1.w;
      a.bf[8] = (__bf16)f2.x;  a.bf[9] = (__bf16)f2.y;
      a.bf[10] = (__bf16)f2.z; a.bf[11] = (__bf16)f2.w;
      a.bf[12] = (__bf16)f3.x; a.bf[13] = (__bf16)f3.y;
      a.bf[14] = (__bf16)f3.z; a.bf[15] = (__bf16)f3.w;
    }
#pragma unroll
    for (int t = 0; t < 4; ++t) {
      Frag b;
      const unsigned short* p =
          Wb + (size_t)(nb + 16 * t + l15) * DMODEL + k + half * 8;
      b.q[0] = *(const uint4*)(p);
      b.q[1] = *(const uint4*)(p + 16);
      acc[t] = wmma_bf16(a, b, acc[t]);
    }
  }

#pragma unroll
  for (int t = 0; t < 4; ++t) {
    const int nt = nb + 16 * t + l15;
    const float bn = bias[nt];
    if (OUTMODE == 0) {
      unsigned short* ob = (unsigned short*)Out;
#pragma unroll
      for (int g = 0; g < 8; ++g) {
        const int row = mb + g + half * 8;
        const int bi = row >> 11, s = row & (SEQ - 1);
        size_t off = (((size_t)(bi * HEADS + (nt >> 6))) * SEQ + s) * HD + (nt & 63);
        ob[off] = f2bf(acc[t][g] + bn);
      }
    } else {
      float* of = (float*)Out;
#pragma unroll
      for (int g = 0; g < 8; ++g) {
        const int row = mb + g + half * 8;
        of[(size_t)row * DMODEL + nt] = acc[t][g] + bn;
      }
    }
  }
}

// ---------------------------------------------------------------------------
// Kernel 3: flash attention, double-buffered & software-pipelined.
// Grid (S/128, H, B), 256 threads (8 waves), 16 query rows per wave,
// key blocks of 64. K tiles move via GLOBAL_LOAD_ASYNC_TO_LDS_B128 into the
// ping-pong buffer while the current tile is being computed; V tiles are
// loaded to registers early and scattered transposed after compute.
// ---------------------------------------------------------------------------
__global__ __launch_bounds__(256) void fa_kernel(
    const unsigned short* __restrict__ Qb,
    const unsigned short* __restrict__ Kb,
    const unsigned short* __restrict__ Vb,
    unsigned short* __restrict__ Ob) {
  __shared__ unsigned short ldsK[2 * 64 * 64];    // 16KB K tiles [buf][key][hd]
  __shared__ unsigned short ldsVt[2 * 64 * 72];   // 18KB V^T [buf][hd][key], stride 72
  __shared__ unsigned short ldsP[8 * 16 * 64];    // 16KB per-wave P (C->A layout)

  const int tid  = threadIdx.x;
  const int lane = tid & 31;
  const int wv   = tid >> 5;
  const int half = lane >> 4;
  const int l15  = lane & 15;

  const int b  = blockIdx.z, h = blockIdx.y;
  const int s0 = blockIdx.x * 128;
  const size_t bh = ((size_t)b * HEADS + h) * SEQ;

  // Q A-fragments for this wave's 16 rows (hd split 0..31 / 32..63)
  Frag aq0, aq1;
  {
    const unsigned short* qrow = Qb + (bh + s0 + wv * 16 + l15) * HD;
    aq0.q[0] = *(const uint4*)(qrow + half * 8);
    aq0.q[1] = *(const uint4*)(qrow + half * 8 + 16);
    aq1.q[0] = *(const uint4*)(qrow + 32 + half * 8);
    aq1.q[1] = *(const uint4*)(qrow + 32 + half * 8 + 16);
  }

  f32x8 o[4];
#pragma unroll
  for (int n = 0; n < 4; ++n) o[n] = (f32x8){};
  float m_[8], l_[8];
#pragma unroll
  for (int g = 0; g < 8; ++g) { m_[g] = -1e30f; l_[g] = 0.0f; }

  const int ldr = tid >> 2;        // 0..63: key row for staging
  const int ldc = (tid & 3) * 16;  // 0,16,32,48: hd column chunk (16 elems)
  // LDS byte offset of this thread's K chunk in buffer 0 (generic ptr low
  // 32 bits = LDS offset per the CDNA5 LDS-aperture mapping, ISA 10.2)
  const unsigned kOff0 = (unsigned)(uintptr_t)&ldsK[ldr * 64 + ldc];

  // ---- prologue: stage tile 0 into buffer 0 ----
  {
    const unsigned short* kg = Kb + (bh + ldr) * HD + ldc;
    asm volatile("global_load_async_to_lds_b128 %0, %1, off"
                 :: "v"(kOff0), "v"(kg) : "memory");
    asm volatile("global_load_async_to_lds_b128 %0, %1, off offset:16"
                 :: "v"(kOff0), "v"(kg) : "memory");
    union { uint4 q; unsigned short s[8]; } v0, v1;
    v0.q = *(const uint4*)(Vb + (bh + ldr) * HD + ldc);
    v1.q = *(const uint4*)(Vb + (bh + ldr) * HD + ldc + 8);
#pragma unroll
    for (int i = 0; i < 8; ++i) {
      ldsVt[(ldc + i) * 72 + ldr]     = v0.s[i];
      ldsVt[(ldc + 8 + i) * 72 + ldr] = v1.s[i];
    }
    asm volatile("s_wait_asynccnt 0x0" ::: "memory");
  }
  __syncthreads();

  for (int kb = 0; kb < SEQ; kb += 64) {
    const int buf  = (kb >> 6) & 1;
    const int nbuf = buf ^ 1;
    const bool more = (kb + 64) < SEQ;

    // ---- issue next tile's loads before computing this one ----
    union { uint4 q; unsigned short s[8]; } nv0, nv1;
    if (more) {
      const unsigned kOffN = kOff0 + (unsigned)(nbuf * 64 * 64 * 2);
      const unsigned short* kg = Kb + (bh + kb + 64 + ldr) * HD + ldc;
      asm volatile("global_load_async_to_lds_b128 %0, %1, off"
                   :: "v"(kOffN), "v"(kg) : "memory");
      asm volatile("global_load_async_to_lds_b128 %0, %1, off offset:16"
                   :: "v"(kOffN), "v"(kg) : "memory");
      nv0.q = *(const uint4*)(Vb + (bh + kb + 64 + ldr) * HD + ldc);
      nv1.q = *(const uint4*)(Vb + (bh + kb + 64 + ldr) * HD + ldc + 8);
    }

    const unsigned short* kC  = ldsK + buf * (64 * 64);
    const unsigned short* vtC = ldsVt + buf * (64 * 72);

    // ---- S = Q K^T for 16 rows x 64 keys (four 16x16 C blocks) ----
    f32x8 c[4];
#pragma unroll
    for (int j = 0; j < 4; ++j) {
      Frag bk;
      const unsigned short* kp = kC + (16 * j + l15) * 64 + half * 8;
      bk.q[0] = *(const uint4*)kp; bk.q[1] = *(const uint4*)(kp + 16);
      c[j] = (f32x8){};
      c[j] = wmma_bf16(aq0, bk, c[j]);
      kp += 32;                                          // hd 32..63
      bk.q[0] = *(const uint4*)kp; bk.q[1] = *(const uint4*)(kp + 16);
      c[j] = wmma_bf16(aq1, bk, c[j]);
    }

    // ---- online softmax ----
    float p[4][8];
#pragma unroll
    for (int g = 0; g < 8; ++g) {
      float s0v = c[0][g] * 0.125f;     // / sqrt(64)
      float s1v = c[1][g] * 0.125f;
      float s2v = c[2][g] * 0.125f;
      float s3v = c[3][g] * 0.125f;
      float t = fmaxf(fmaxf(s0v, s1v), fmaxf(s2v, s3v));
      t = xormax16(t);
      const float mnew  = fmaxf(m_[g], t);
      const float alpha = exp2f((m_[g] - mnew) * 1.4426950408889634f);
      const float e0 = exp2f((s0v - mnew) * 1.4426950408889634f);
      const float e1 = exp2f((s1v - mnew) * 1.4426950408889634f);
      const float e2 = exp2f((s2v - mnew) * 1.4426950408889634f);
      const float e3 = exp2f((s3v - mnew) * 1.4426950408889634f);
      float rs = (e0 + e1) + (e2 + e3);
      rs = xorsum16(rs);
      l_[g] = l_[g] * alpha + rs;
      m_[g] = mnew;
      p[0][g] = e0; p[1][g] = e1; p[2][g] = e2; p[3][g] = e3;
#pragma unroll
      for (int n = 0; n < 4; ++n) o[n][g] *= alpha;
    }

    // ---- P: f32 C-layout -> bf16 A-layout via per-wave LDS bounce ----
    unsigned short* pw = &ldsP[wv * 1024];
#pragma unroll
    for (int j = 0; j < 4; ++j)
#pragma unroll
      for (int g = 0; g < 8; ++g) {
        const int row = g + half * 8;
        pw[row * 64 + 16 * j + l15] = f2bf(p[j][g]);
      }
    asm volatile("s_wait_dscnt 0x0" ::: "memory");  // same-wave LDS RAW
    Frag ap0, ap1;
    {
      const unsigned short* pp = pw + l15 * 64 + half * 8;
      ap0.q[0] = *(const uint4*)pp;
      ap0.q[1] = *(const uint4*)(pp + 16);
      ap1.q[0] = *(const uint4*)(pp + 32);
      ap1.q[1] = *(const uint4*)(pp + 48);
    }

    // ---- O += P @ V ----
#pragma unroll
    for (int n = 0; n < 4; ++n) {
      Frag bvf;
      const unsigned short* vp = vtC + (n * 16 + l15) * 72 + half * 8;
      bvf.q[0] = *(const uint4*)vp;
      bvf.q[1] = *(const uint4*)(vp + 16);
      o[n] = wmma_bf16(ap0, bvf, o[n]);
      vp += 32;                                          // keys 32..63
      bvf.q[0] = *(const uint4*)vp;
      bvf.q[1] = *(const uint4*)(vp + 16);
      o[n] = wmma_bf16(ap1, bvf, o[n]);
    }

    // ---- scatter next V tile (transposed) after compute ----
    if (more) {
      unsigned short* vtN = ldsVt + nbuf * (64 * 72);
#pragma unroll
      for (int i = 0; i < 8; ++i) {
        vtN[(ldc + i) * 72 + ldr]     = nv0.s[i];
        vtN[(ldc + 8 + i) * 72 + ldr] = nv1.s[i];
      }
    }
    asm volatile("s_wait_asynccnt 0x0" ::: "memory");
    __syncthreads();
  }

  // epilogue: O / l -> bf16, merged-head layout [b, s, h*64+d]
  unsigned short* ob = Ob + ((size_t)b * SEQ) * DMODEL + (size_t)h * HD;
#pragma unroll
  for (int n = 0; n < 4; ++n) {
#pragma unroll
    for (int g = 0; g < 8; ++g) {
      const int srow = s0 + wv * 16 + g + half * 8;
      ob[(size_t)srow * DMODEL + n * 16 + l15] = f2bf(o[n][g] / l_[g]);
    }
  }
}

// ---------------------------------------------------------------------------
extern "C" void kernel_launch(void* const* d_in, const int* in_sizes, int n_in,
                              void* d_out, int out_size, void* d_ws, size_t ws_size,
                              hipStream_t stream) {
  (void)in_sizes; (void)n_in; (void)out_size; (void)ws_size;
  const float* q  = (const float*)d_in[0];
  const float* k  = (const float*)d_in[1];
  const float* v  = (const float*)d_in[2];
  // d_in[3] = mask: reference discards the masked_fill result -> unused
  const float* Wq = (const float*)d_in[4];
  const float* bq = (const float*)d_in[5];
  const float* Wk = (const float*)d_in[6];
  const float* bk = (const float*)d_in[7];
  const float* Wv = (const float*)d_in[8];
  const float* bv = (const float*)d_in[9];
  const float* Wo = (const float*)d_in[10];
  const float* bo = (const float*)d_in[11];

  unsigned short* ws   = (unsigned short*)d_ws;
  unsigned short* Wq_b = ws;
  unsigned short* Wk_b = Wq_b + 512 * 512;
  unsigned short* Wv_b = Wk_b + 512 * 512;
  unsigned short* Wo_b = Wv_b + 512 * 512;
  unsigned short* Qb   = Wo_b + 512 * 512;            // [B,H,S,hd]
  unsigned short* Kb   = Qb + 4u * HEADS * SEQ * HD;
  unsigned short* Vb   = Kb + 4u * HEADS * SEQ * HD;
  unsigned short* Ab   = Vb + 4u * HEADS * SEQ * HD;  // merged-head attn out

  wcvt_kernel<<<dim3(256, 1, 4), 256, 0, stream>>>(Wq, Wk, Wv, Wo, Wq_b);

  dim3 ggrid(128, 4, 1);
  gemm_kernel<0, 0><<<ggrid, 256, 0, stream>>>(q, Wq_b, bq, Qb);
  gemm_kernel<0, 0><<<ggrid, 256, 0, stream>>>(k, Wk_b, bk, Kb);
  gemm_kernel<0, 0><<<ggrid, 256, 0, stream>>>(v, Wv_b, bv, Vb);

  fa_kernel<<<dim3(SEQ / 128, HEADS, 4), 256, 0, stream>>>(Qb, Kb, Vb, Ab);

  gemm_kernel<1, 1><<<ggrid, 256, 0, stream>>>(Ab, Wo_b, bo, d_out);
}
